// OutlierAwareLinear_44409961840721
// MI455X (gfx1250) — 8-bit matrix operand order repaired and checked by reading, compiled, not run on hardware
//
#include <hip/hip_runtime.h>
#include <hip/hip_bf16.h>

#define TOK   256
#define IN_F  4096
#define OUT_F 4096
#define NOUT  32
#define KCH   64
#define MT    4
#define APITCH 80

typedef int v8i __attribute__((ext_vector_type(8)));

#if __has_builtin(__builtin_amdgcn_global_load_async_to_lds_b128) && \
    __has_builtin(__builtin_amdgcn_s_wait_asynccnt)
#define USE_ASYNC_LDS 1
typedef __attribute__((__vector_size__(4 * sizeof(int)))) int vec4i;
typedef __attribute__((address_space(1))) vec4i* gvec_t;
typedef __attribute__((address_space(3))) vec4i* lvec_t;
#endif

__device__ __forceinline__ unsigned pack4(int4 v) {
    unsigned lo = __builtin_amdgcn_perm((unsigned)v.y, (unsigned)v.x, 0x0C0C0400u);
    unsigned hi = __builtin_amdgcn_perm((unsigned)v.w, (unsigned)v.z, 0x0C0C0400u);
    return __builtin_amdgcn_perm(hi, lo, 0x05040100u);
}

__global__ void __launch_bounds__(256) k_quant(
    const float* __restrict__ x, const int* __restrict__ oidx,
    signed char* __restrict__ xq, float* __restrict__ xs,
    float* __restrict__ xout)
{
    __shared__ unsigned char mask[IN_F];
    __shared__ float red[256];
    const int row = blockIdx.x;
    const int t   = threadIdx.x;

    #pragma unroll
    for (int i = 0; i < 16; ++i) mask[t * 16 + i] = 1;
    __syncthreads();
    if (t < NOUT) mask[oidx[t]] = 0;
    __syncthreads();

    const float4* xr = (const float4*)(x + (size_t)row * IN_F);
    float mval[16];
    float am = 0.f;
    #pragma unroll
    for (int j = 0; j < 4; ++j) {
        float4 f = xr[t * 4 + j];
        float e[4] = {f.x, f.y, f.z, f.w};
        #pragma unroll
        for (int c = 0; c < 4; ++c) {
            int k = t * 16 + j * 4 + c;
            float m = mask[k] ? e[c] : 0.f;
            mval[j * 4 + c] = m;
            am = fmaxf(am, fabsf(m));
        }
    }

    red[t] = am;
    __syncthreads();
    for (int s = 128; s > 0; s >>= 1) {
        if (t < s) red[t] = fmaxf(red[t], red[t + s]);
        __syncthreads();
    }
    const float scale = red[0];
    const float inv = 127.f / fmaxf(scale, 1e-8f);

    unsigned pk[4];
    #pragma unroll
    for (int j = 0; j < 4; ++j) {
        unsigned p = 0;
        #pragma unroll
        for (int c = 0; c < 4; ++c) {
            float q = rintf(mval[j * 4 + c] * inv);
            q = fminf(fmaxf(q, -127.f), 127.f);
            int qi = (int)q;
            p |= ((unsigned)(qi & 255)) << (8 * c);
        }
        pk[j] = p;
    }
    ((uint4*)(xq + (size_t)row * IN_F))[t] = make_uint4(pk[0], pk[1], pk[2], pk[3]);

    if (t == 0) xs[row] = scale;
    if (t < NOUT) xout[row * NOUT + t] = x[(size_t)row * IN_F + oidx[t]];
}

__global__ void __launch_bounds__(256) k_outlier(
    const float* __restrict__ xout, const float* __restrict__ outw,
    const float* __restrict__ bias, float* __restrict__ out)
{
    __shared__ __align__(16) float sxo[TOK * NOUT];
    const int t = threadIdx.x;
    {
        const float4* src = (const float4*)xout;
        float4* dst = (float4*)sxo;
        #pragma unroll
        for (int i = 0; i < 8; ++i) dst[t + 256 * i] = src[t + 256 * i];
    }
    __syncthreads();

    const int col = blockIdx.x * 256 + t;
    float w[NOUT];
    const float4* wr = (const float4*)(outw + (size_t)col * NOUT);
    #pragma unroll
    for (int i = 0; i < 8; ++i) {
        float4 f = wr[i];
        w[4*i+0] = f.x; w[4*i+1] = f.y; w[4*i+2] = f.z; w[4*i+3] = f.w;
    }
    const float b = bias[col];

    for (int m = 0; m < TOK; ++m) {
        const float* xm = sxo + m * NOUT;
        float s = b;
        #pragma unroll
        for (int o = 0; o < NOUT; ++o) s = fmaf(xm[o], w[o], s);
        out[(size_t)m * OUT_F + col] = s;
    }
}

__global__ void __launch_bounds__(256, 1) k_gemm(
    const signed char* __restrict__ xq, const float* __restrict__ xs,
    const int* __restrict__ w, const float* __restrict__ wscale,
    float* __restrict__ out)
{
    __shared__ __align__(16) signed char sA[2][64][APITCH];
    __shared__ float sXs[64];

    const int t    = threadIdx.x;
    const int lane = t & 31;
    const int wv   = t >> 5;
    const int m0   = blockIdx.y * 64;
    const int n0   = blockIdx.x * 128 + wv * 16;
    const int col  = n0 + (lane & 15);

    if (t < 64) sXs[t] = xs[m0 + t] * (1.f / 127.f);

    v8i acc[MT];
    #pragma unroll
    for (int mt = 0; mt < MT; ++mt) acc[mt] = (v8i){0,0,0,0,0,0,0,0};

    const int* wrow = w + (size_t)col * IN_F + ((lane >> 4) * 16);
    const int koffA = (lane & 16) ? 16 : 0;

    for (int kc = 0; kc < IN_F / KCH; ++kc) {
        const int kb = kc * KCH;
        {
            const int r = t >> 2;
            const int c = (t & 3) * 16;
            const signed char* gsrc = xq + (size_t)(m0 + r) * IN_F + kb + c;
#ifdef USE_ASYNC_LDS
            gvec_t gp = (gvec_t)(size_t)gsrc;
            lvec_t lp = (lvec_t)(unsigned)(size_t)(&sA[kc & 1][r][c]);
            __builtin_amdgcn_global_load_async_to_lds_b128(gp, lp, 0, 0);
#else
            *(uint4*)&sA[kc & 1][r][c] = *(const uint4*)gsrc;
#endif
        }
#ifdef USE_ASYNC_LDS
        __builtin_amdgcn_s_wait_asynccnt(0);
#endif
        __syncthreads();

        if (kc + 1 < IN_F / KCH)
            __builtin_prefetch((const void*)(wrow + kb + KCH), 0, 1);

        const int4* wp = (const int4*)(wrow + kb);
        int4 w0 = wp[0], w1 = wp[1], w2 = wp[2],  w3 = wp[3];
        int4 w4 = wp[8], w5 = wp[9], w6 = wp[10], w7 = wp[11];
        v8i bfr;
        bfr[0] = (int)pack4(w0); bfr[1] = (int)pack4(w1);
        bfr[2] = (int)pack4(w2); bfr[3] = (int)pack4(w3);
        bfr[4] = (int)pack4(w4); bfr[5] = (int)pack4(w5);
        bfr[6] = (int)pack4(w6); bfr[7] = (int)pack4(w7);

        #pragma unroll
        for (int mt = 0; mt < MT; ++mt) {
            const int r = mt * 16 + (lane & 15);
            const signed char* ap = &sA[kc & 1][r][koffA];
            int2 a01 = *(const int2*)(ap);
            int2 a23 = *(const int2*)(ap + 8);
            int2 a45 = *(const int2*)(ap + 32);
            int2 a67 = *(const int2*)(ap + 40);
            v8i afr = { a01.x, a01.y, a23.x, a23.y, a45.x, a45.y, a67.x, a67.y };
            acc[mt] = __builtin_amdgcn_wmma_i32_16x16x64_iu8(
                 true, afr,  true, bfr, acc[mt],
                 false,  false);
        }
    }
    __syncthreads();

    const float wsc = wscale[col] * (1.f / 127.f);
    const int mlo = ((lane >> 4) & 1) * 8;
    #pragma unroll
    for (int mt = 0; mt < MT; ++mt) {
        #pragma unroll
        for (int j = 0; j < 8; ++j) {
            const int m = mt * 16 + mlo + j;
            const float v = (float)acc[mt][j] * sXs[m] * wsc;
            float* o = out + (size_t)(m0 + m) * OUT_F + col;
            *o = *o + v;
        }
    }
}

extern "C" void kernel_launch(void* const* d_in, const int* in_sizes, int n_in,
                              void* d_out, int out_size, void* d_ws, size_t ws_size,
                              hipStream_t stream) {
    const float* x     = (const float*)d_in[0];
    const int*   wq    = (const int*)  d_in[1];
    const float* wsc   = (const float*)d_in[2];
    const int*   oidx  = (const int*)  d_in[3];
    const float* outw  = (const float*)d_in[4];
    const float* bias  = (const float*)d_in[5];
    float*       out   = (float*)d_out;

    signed char* xq   = (signed char*)d_ws;
    float*       xs   = (float*)((char*)d_ws + (size_t)TOK * IN_F);
    float*       xout = xs + 256;

    k_quant<<<TOK, 256, 0, stream>>>(x, oidx, xq, xs, xout);
    k_outlier<<<OUT_F / 256, 256, 0, stream>>>(xout, outw, bias, out);
    dim3 g(OUT_F / 128, TOK / 64);
    k_gemm<<<g, 256, 0, stream>>>(xq, xs, wq, wsc, out);
}
